// UpdateModule_6640019440395
// MI455X (gfx1250) — matrix operand order repaired and checked by reading, compiled, not run on hardware
//
#include <hip/hip_runtime.h>
#include <hip/hip_bf16.h>
#include <math.h>

// ---------------- problem constants ----------------
constexpr int BB   = 64;
constexpr int SS   = 32;
constexpr int SYM  = 256;
constexpr int HID  = 256;
constexpr int ENT  = 128;
constexpr int ROLE = 64;
constexpr int E_   = ENT;    // 128
constexpr int R_   = ROLE;   // 64
constexpr int F_   = ENT;    // 128
constexpr int FC   = 4;      // f-columns per tile
constexpr int NCOL = R_ * FC;            // 256 packed (r,fc)
constexpr int TPR_PER_B = E_ * R_ * F_;  // 2^20

// padded LDS strides (floats) — multiples of 4 for b128 alignment, bank-spread
constexpr int TSTR  = E_ + 4;   // 132: Tt2[ncol][e]
constexpr int USTR  = E_ + 4;   // 132: u[s][e]
constexpr int UTSTR = SS + 4;   // 36 : uT[e][s]
constexpr int PSTR  = SS + 4;   // 36 : Pt/OPt[ncol][s]
constexpr int RSTR  = R_ + 1;   // 65 : r*[s][r]

typedef __attribute__((ext_vector_type(8)))  float  v8f;
typedef __attribute__((ext_vector_type(16))) __bf16 v16bf;
typedef __attribute__((ext_vector_type(8)))  __bf16 v8bf;

#ifndef __has_builtin
#define __has_builtin(x) 0
#endif
#if __has_builtin(__builtin_amdgcn_global_load_async_to_lds_b32)
#define HAVE_ASYNC_LDS 1
#else
#define HAVE_ASYNC_LDS 0
#endif
#if __has_builtin(__builtin_amdgcn_tensor_load_to_lds) && \
    __has_builtin(__builtin_amdgcn_s_wait_tensorcnt)
#define HAVE_TDM 1
#else
#define HAVE_TDM 0
#endif

// ---- operand builders: two 8-contiguous-f32 blocks -> v16bf (contiguous k order, same for A and B) ----
// A 16x32: lane<16 K=0..15 ; lane>=16 K=16..31  (16 contiguous)
// B 32x16: lane<16 K=0..15 ; lane>=16 K=16..31  (16 contiguous)
__device__ __forceinline__ v16bf pack_ab(const float* p0, const float* p1) {
  const float4 q0 = *(const float4*)p0;
  const float4 q1 = *(const float4*)(p0 + 4);
  const float4 q2 = *(const float4*)p1;
  const float4 q3 = *(const float4*)(p1 + 4);
  v16bf r;
  r[0]=(__bf16)q0.x; r[1]=(__bf16)q0.y; r[2]=(__bf16)q0.z; r[3]=(__bf16)q0.w;
  r[4]=(__bf16)q1.x; r[5]=(__bf16)q1.y; r[6]=(__bf16)q1.z; r[7]=(__bf16)q1.w;
  r[8]=(__bf16)q2.x; r[9]=(__bf16)q2.y; r[10]=(__bf16)q2.z; r[11]=(__bf16)q2.w;
  r[12]=(__bf16)q3.x; r[13]=(__bf16)q3.y; r[14]=(__bf16)q3.z; r[15]=(__bf16)q3.w;
  return r;
}
__device__ __forceinline__ v16bf pack_bf(const __bf16* p0, const __bf16* p1) {
  const v8bf lo = *(const v8bf*)p0;
  const v8bf hi = *(const v8bf*)p1;
  v16bf r;
#pragma unroll
  for (int i = 0; i < 8; ++i) { r[i] = lo[i]; r[8 + i] = hi[i]; }
  return r;
}

// ---- CDNA5 async copy global->LDS (ASYNCcnt) ----
__device__ __forceinline__ void async_ld_b32(const float* g, float* l) {
#if HAVE_ASYNC_LDS
  __builtin_amdgcn_global_load_async_to_lds_b32((int*)g, (int*)l, 0, 0);
#endif
}
__device__ __forceinline__ void wait_async0() {
#if HAVE_ASYNC_LDS
#if __has_builtin(__builtin_amdgcn_s_wait_asynccnt)
  __builtin_amdgcn_s_wait_asynccnt(0);
#else
  asm volatile("s_wait_asynccnt 0" ::: "memory");
#endif
#endif
}

// ---- CDNA5 TDM: 2D tile load with LDS row padding (TENSORcnt) ----
#if HAVE_TDM
typedef unsigned int uint32x4 __attribute__((ext_vector_type(4)));
typedef int          int32x4  __attribute__((ext_vector_type(4)));
typedef int          int32x8  __attribute__((ext_vector_type(8)));
// dim0 dwords per row, rows rows; pad 'pamt' dwords (code) every 2^(pint+1) dwords
__device__ __forceinline__ void tdm_load_2d(const float* gsrc, const float* ldst,
                                            int dim0, int rows, int pint, int pamt,
                                            int stride0) {
  const unsigned long long ga = (unsigned long long)(uintptr_t)gsrc;
  const unsigned int       la = (unsigned int)(uintptr_t)ldst;
  uint32x4 g0;
  g0[0] = 1u;                                      // count=1, user mode
  g0[1] = la;                                      // lds_addr
  g0[2] = (unsigned int)ga;                        // global_addr[31:0]
  g0[3] = (unsigned int)((ga >> 32) & 0x1FFFFFFull) | (2u << 30); // addr[56:32]|type=2
  int32x8 g1;
  g1[0] = (int)((2u << 16) | (1u << 20) |          // data_size=4B, pad_enable
                ((unsigned)pint << 22) | ((unsigned)pamt << 25));
  g1[1] = (int)((unsigned)(dim0 & 0xFFFF) << 16);  // tensor_dim0 lo16 (bits 63:48)
  g1[2] = (int)(((unsigned)dim0 >> 16) |           // tensor_dim0 hi16
                ((unsigned)(rows & 0xFFFF) << 16));// tensor_dim1 lo16
  g1[3] = (int)(((unsigned)rows >> 16) |
                ((unsigned)(dim0 & 0xFFFF) << 16));// tile_dim0 = dim0
  g1[4] = (int)(rows & 0xFFFF);                    // tile_dim1 = rows, tile_dim2=0
  g1[5] = stride0;                                 // tensor_dim0_stride lo32
  g1[6] = 0; g1[7] = 0;
  const int32x4 z4 = {0, 0, 0, 0};
  const int32x8 z8 = {0, 0, 0, 0, 0, 0, 0, 0};
  __builtin_amdgcn_tensor_load_to_lds(g0, g1, z4, z4, z8, 0);
}
#endif

// =====================================================================
// Kernel 1: fused 5-head MLP
// =====================================================================
__global__ __launch_bounds__(256) void mlp_kernel(
    const float* __restrict__ x,
    const float* __restrict__ eW1, const float* __restrict__ eb1,
    const float* __restrict__ eW2, const float* __restrict__ eb2,
    const float* __restrict__ rW1, const float* __restrict__ rb1,
    const float* __restrict__ rW2, const float* __restrict__ rb2,
    float* __restrict__ e1, float* __restrict__ e2,
    float* __restrict__ r1, float* __restrict__ r2, float* __restrict__ r3)
{
  const int head = blockIdx.y;
  const int row0 = blockIdx.x * 32;

  const float *W1, *b1, *W2, *b2; float* out; int O;
  if (head < 2) {
    W1 = eW1 + head * SYM * HID; b1 = eb1 + head * HID;
    W2 = eW2 + head * HID * ENT; b2 = eb2 + head * ENT;
    out = (head == 0) ? e1 : e2; O = ENT;
  } else {
    const int h = head - 2;
    W1 = rW1 + h * SYM * HID; b1 = rb1 + h * HID;
    W2 = rW2 + h * HID * ROLE; b2 = rb2 + h * ROLE;
    out = (h == 0) ? r1 : ((h == 1) ? r2 : r3); O = ROLE;
  }

  __shared__ __bf16 Xs[32][SYM];
  __shared__ __bf16 Hs[32][HID];

  const int tid  = threadIdx.x;
  const int wave = tid >> 5;
  const int lane = tid & 31;
  const int n    = lane & 15;
  const int hi8  = (lane < 16) ? 0 : 8;
  const int baseA = (lane < 16) ? 0 : 16;
  const int baseB = (lane < 16) ? 0 : 16;

  for (int i = tid; i < 32 * SYM; i += 256)
    Xs[i / SYM][i % SYM] = (__bf16)x[(size_t)(row0 + i / SYM) * SYM + (i % SYM)];
  __syncthreads();

  for (int t = wave; t < 2 * (HID / 16); t += 8) {
    const int mt = t >> 4, nt = t & 15;
    const int m = mt * 16 + n;
    v8f acc = {};
    for (int k0 = 0; k0 < SYM; k0 += 32) {
      const v16bf a = pack_bf(&Xs[m][k0 + baseA], &Xs[m][k0 + 8 + baseA]);
      v16bf bm;
#pragma unroll
      for (int i = 0; i < 16; ++i)
        bm[i] = (__bf16)W1[(size_t)(k0 + baseB + i) * HID + nt * 16 + n];
      acc = __builtin_amdgcn_wmma_f32_16x16x32_bf16(false, a, false, bm,
                                                    (short)0, acc, false, false);
    }
#pragma unroll
    for (int v = 0; v < 8; ++v)
      Hs[mt * 16 + v + hi8][nt * 16 + n] =
          (__bf16)fmaxf(acc[v] + b1[nt * 16 + n], 0.0f);
  }
  __syncthreads();

  for (int t = wave; t < 2 * (O / 16); t += 8) {
    const int mt = t >> 4, nt = t & 15;
    const int m = mt * 16 + n;
    v8f acc = {};
    for (int k0 = 0; k0 < HID; k0 += 32) {
      const v16bf a = pack_bf(&Hs[m][k0 + baseA], &Hs[m][k0 + 8 + baseA]);
      v16bf bm;
#pragma unroll
      for (int i = 0; i < 16; ++i)
        bm[i] = (__bf16)W2[(size_t)(k0 + baseB + i) * O + nt * 16 + n];
      acc = __builtin_amdgcn_wmma_f32_16x16x32_bf16(false, a, false, bm,
                                                    (short)0, acc, false, false);
    }
#pragma unroll
    for (int v = 0; v < 8; ++v)
      out[(size_t)(row0 + mt * 16 + v + hi8) * O + nt * 16 + n] =
          acc[v] + b2[nt * 16 + n];
  }
}

// =====================================================================
// Kernel 2: one TPR layer, fused in LDS per (batch, f-tile).
// =====================================================================
__global__ __launch_bounds__(256) void layer_kernel(
    const float* __restrict__ Tin, float* __restrict__ Tout,
    const float* __restrict__ e1g, const float* __restrict__ e2g,
    const float* __restrict__ r1g, const float* __restrict__ r2g,
    const float* __restrict__ r3g,
    const float* __restrict__ sigma, float* __restrict__ nacc,
    int layer0)
{
  extern __shared__ float sm[];
  float* Tt2 = sm;                         // [NCOL][TSTR]      33792 f
  float* e1s = Tt2 + NCOL * TSTR;          // [S][USTR]          4224
  float* e2s = e1s + SS * USTR;            //                    4224
  float* e1T = e2s + SS * USTR;            // [E][UTSTR]         4608
  float* e2T = e1T + E_ * UTSTR;           //                    4608
  float* r1s = e2T + E_ * UTSTR;           // [S][RSTR]          2080
  float* r2s = r1s + SS * RSTR;
  float* r3s = r2s + SS * RSTR;
  float* Pt  = r3s + SS * RSTR;            // [NCOL][PSTR]       9216 (reused as OPt)
  float* wh  = Pt  + NCOL * PSTR;          // [S*FC]              128
  float* gv  = wh  + SS * FC;              //                     128

  const int b   = blockIdx.y;
  const int f0  = blockIdx.x * FC;
  const int tid = threadIdx.x;
  const int wave = tid >> 5;
  const int lane = tid & 31;
  const int n    = lane & 15;
  const int hi8  = (lane < 16) ? 0 : 8;
  const int baseA = (lane < 16) ? 0 : 16;  // A-operand K base
  const int baseB = (lane < 16) ? 0 : 16;  // B-operand K base

  const float sig    = sigma[b];
  const float invsig = 1.0f / sig;

  // ---- stage vectors (TDM with LDS padding if available) ----
#if HAVE_TDM
  if (tid == 0) {
    tdm_load_2d(e1g + (size_t)b * SS * E_, e1s, E_, SS, 6, 3, E_); // pad 4/128 dw
    tdm_load_2d(e2g + (size_t)b * SS * E_, e2s, E_, SS, 6, 3, E_);
    tdm_load_2d(r1g + (size_t)b * SS * R_, r1s, R_, SS, 5, 0, R_); // pad 1/64 dw
    tdm_load_2d(r2g + (size_t)b * SS * R_, r2s, R_, SS, 5, 0, R_);
    tdm_load_2d(r3g + (size_t)b * SS * R_, r3s, R_, SS, 5, 0, R_);
  }
#else
  for (int i = tid; i < SS * E_; i += 256) {
    const int s = i >> 7, e = i & 127;
    e1s[s * USTR + e] = e1g[(size_t)b * SS * E_ + i];
    e2s[s * USTR + e] = e2g[(size_t)b * SS * E_ + i];
  }
  for (int i = tid; i < SS * R_; i += 256) {
    const int s = i >> 6, r = i & 63;
    r1s[s * RSTR + r] = r1g[(size_t)b * SS * R_ + i];
    r2s[s * RSTR + r] = r2g[(size_t)b * SS * R_ + i];
    r3s[s * RSTR + r] = r3g[(size_t)b * SS * R_ + i];
  }
#endif
  // transposed u (always plain; TDM cannot transpose)
  for (int i = tid; i < SS * E_; i += 256) {
    const int s = i >> 7, e = i & 127;
    e1T[e * UTSTR + s] = e1g[(size_t)b * SS * E_ + i];
    e2T[e * UTSTR + s] = e2g[(size_t)b * SS * E_ + i];
  }

  // ---- stage T tile transposed: Tt2[(r*FC+fc)*TSTR + e] ----
  const float* Tsrc = layer0 ? Tin : (Tin + (size_t)b * TPR_PER_B);
  for (int idx = tid; idx < E_ * R_; idx += 256) {
    const int r = idx & (R_ - 1), e = idx >> 6;     // row (e*R+r) == idx
    const float* g = Tsrc + (size_t)idx * F_ + f0;
    float* l = Tt2 + (r * FC) * TSTR + e;
#if HAVE_ASYNC_LDS
    async_ld_b32(g + 0, l);
    async_ld_b32(g + 1, l + TSTR);
    async_ld_b32(g + 2, l + 2 * TSTR);
    async_ld_b32(g + 3, l + 3 * TSTR);
#else
    const float4 v = *(const float4*)g;
    l[0] = v.x; l[TSTR] = v.y; l[2 * TSTR] = v.z; l[3 * TSTR] = v.w;
#endif
  }
  wait_async0();
#if HAVE_TDM
  if (tid == 0) __builtin_amdgcn_s_wait_tensorcnt(0);
#endif
  __syncthreads();

  // steps: 0 write(u=e1,v=r1,g=e2-wh) 1 move(u=e1,v=r2,g=wh-mh) 2 backlink(u=e2,v=r3,g=e1-bh)
  for (int step = 0; step < 3; ++step) {
    const float* u  = (step == 2) ? e2s : e1s;
    const float* uT = (step == 2) ? e2T : e1T;
    const float* vv = (step == 0) ? r1s : ((step == 1) ? r2s : r3s);

    // ---- hat stage 1: Pt[ncol][s] = sum_e u[s,e]*Tt2[ncol][e]  (M=32,N=256,K=128)
    for (int t = wave; t < 2 * (NCOL / 16); t += 8) {
      const int mt = t >> 4, nt = t & 15;
      const int m = mt * 16 + n;            // s row
      const int col = nt * 16 + n;
      v8f acc = {};
      for (int k0 = 0; k0 < E_; k0 += 32) {
        const float* pa = u + m * USTR + k0 + baseA;
        const v16bf a = pack_ab(pa, pa + 8);
        const float* pb = Tt2 + col * TSTR + k0 + baseB;
        const v16bf bm = pack_ab(pb, pb + 8);
        acc = __builtin_amdgcn_wmma_f32_16x16x32_bf16(false, a, false, bm,
                                                      (short)0, acc, false, false);
      }
      float* pc = Pt + col * PSTR + mt * 16 + hi8;
      *(float4*)pc       = make_float4(acc[0], acc[1], acc[2], acc[3]);
      *(float4*)(pc + 4) = make_float4(acc[4], acc[5], acc[6], acc[7]);
    }
    __syncthreads();

    // ---- hat stage 2: h[s,fc] = sig * sum_r vv[s,r]*Pt[(r*FC+fc)][s]
    if (tid < SS * FC) {
      const int s = tid & 31, fc = tid >> 5;
      float acc = 0.0f;
#pragma unroll 8
      for (int r = 0; r < R_; ++r) acc += vv[s * RSTR + r] * Pt[(r * FC + fc) * PSTR + s];
      const float h = sig * acc;
      const int idx = s * FC + fc;
      float g;
      if (step == 0)      { wh[idx] = h; g = e2s[s * USTR + f0 + fc] - h; }
      else if (step == 1) { g = wh[idx] - h; }
      else                { g = e1s[s * USTR + f0 + fc] - h; }
      gv[idx] = g * invsig;
    }
    __syncthreads();

    // ---- materialize update-B transposed: OPt[ncol][s] = vv[s,r]*gv[s,fc]
    for (int i = tid; i < NCOL * SS; i += 256) {
      const int s = i & 31, col = i >> 5;
      Pt[col * PSTR + s] = vv[s * RSTR + (col >> 2)] * gv[s * FC + (col & 3)];
    }
    __syncthreads();

    // ---- update: Tt2[ncol][e] += sum_s uT[e,s]*OPt[ncol][s]  (M=128,N=256,K=32)
    for (int t = wave; t < 8 * (NCOL / 16); t += 8) {
      const int mt = t >> 4, nt = t & 15;
      const int m = mt * 16 + n;            // e row
      const int col = nt * 16 + n;
      float* pc = Tt2 + col * TSTR + mt * 16 + hi8;
      const float4 c0 = *(const float4*)pc;
      const float4 c1 = *(const float4*)(pc + 4);
      v8f c = {c0.x, c0.y, c0.z, c0.w, c1.x, c1.y, c1.z, c1.w};
      const float* pa = uT + m * UTSTR + baseA;
      const v16bf a = pack_ab(pa, pa + 8);
      const float* pb = Pt + col * PSTR + baseB;
      const v16bf bm = pack_ab(pb, pb + 8);
      c = __builtin_amdgcn_wmma_f32_16x16x32_bf16(false, a, false, bm,
                                                  (short)0, c, false, false);
      *(float4*)pc       = make_float4(c[0], c[1], c[2], c[3]);
      *(float4*)(pc + 4) = make_float4(c[4], c[5], c[6], c[7]);
    }
    __syncthreads();
  }

  // ---- write back + per-batch ||T*||^2 partial ----
  float ss = 0.0f;
  for (int idx = tid; idx < E_ * R_; idx += 256) {
    const int r = idx & (R_ - 1), e = idx >> 6;
    const float* l = Tt2 + (r * FC) * TSTR + e;
    const float4 v = make_float4(l[0], l[TSTR], l[2 * TSTR], l[3 * TSTR]);
    ss += v.x * v.x + v.y * v.y + v.z * v.z + v.w * v.w;
    *(float4*)(Tout + (size_t)b * TPR_PER_B + (size_t)idx * F_ + f0) = v;
  }
  for (int off = 16; off > 0; off >>= 1) ss += __shfl_down(ss, off);
  __shared__ float red[8];
  if (lane == 0) red[wave] = ss;
  __syncthreads();
  if (tid == 0) {
    float tot = 0.0f;
    for (int w = 0; w < 8; ++w) tot += red[w];
    atomicAdd(nacc + b, tot);
  }
}

// =====================================================================
// Tiny per-batch kernels
// =====================================================================
__global__ void init_sigma_kernel(float* sigma, float* nacc) {
  const int b = threadIdx.x;
  if (b < BB) { sigma[b] = 1.0f; nacc[b] = 0.0f; }
}

__global__ void sigma_kernel(float* sigma, float* nacc) {
  const int b = threadIdx.x;
  if (b < BB) {
    const float s   = sigma[b];
    const float nrm = s * sqrtf(nacc[b]);
    sigma[b] = s / (fmaxf(nrm - 1.0f, 0.0f) + 1.0f);
    nacc[b]  = 0.0f;
  }
}

__global__ __launch_bounds__(256) void final_scale_kernel(
    float* __restrict__ T, const float* __restrict__ sigma)
{
  const size_t i = ((size_t)blockIdx.x * 256 + threadIdx.x) * 4;
  const float  s = sigma[i >> 20];
  float4* p = (float4*)(T + i);
  float4 v = *p;
  v.x *= s; v.y *= s; v.z *= s; v.w *= s;
  *p = v;
}

// =====================================================================
// Host entry
// =====================================================================
extern "C" void kernel_launch(void* const* d_in, const int* in_sizes, int n_in,
                              void* d_out, int out_size, void* d_ws, size_t ws_size,
                              hipStream_t stream) {
  const float* x   = (const float*)d_in[0];
  const float* eW1 = (const float*)d_in[1];
  const float* eb1 = (const float*)d_in[2];
  const float* eW2 = (const float*)d_in[3];
  const float* eb2 = (const float*)d_in[4];
  const float* rW1 = (const float*)d_in[5];
  const float* rb1 = (const float*)d_in[6];
  const float* rW2 = (const float*)d_in[7];
  const float* rb2 = (const float*)d_in[8];
  const float* fwm = (const float*)d_in[9];
  float* T = (float*)d_out;

  float* ws = (float*)d_ws;
  float* e1    = ws;
  float* e2    = e1 + BB * SS * ENT;
  float* r1    = e2 + BB * SS * ENT;
  float* r2    = r1 + BB * SS * ROLE;
  float* r3    = r2 + BB * SS * ROLE;
  float* sigma = r3 + BB * SS * ROLE;
  float* nacc  = sigma + BB;

  init_sigma_kernel<<<1, 64, 0, stream>>>(sigma, nacc);

  mlp_kernel<<<dim3((BB * SS) / 32, 5), 256, 0, stream>>>(
      x, eW1, eb1, eW2, eb2, rW1, rb1, rW2, rb2, e1, e2, r1, r2, r3);

  const size_t layer_smem =
      (size_t)(NCOL * TSTR + 2 * SS * USTR + 2 * E_ * UTSTR + 3 * SS * RSTR +
               NCOL * PSTR + 2 * SS * FC) * sizeof(float);   // ~263 KB / 320 KB

  for (int layer = 0; layer < 3; ++layer) {
    layer_kernel<<<dim3(F_ / FC, BB), 256, layer_smem, stream>>>(
        (layer == 0) ? fwm : T, T, e1, e2, r1, r2, r3, sigma, nacc, layer == 0);
    sigma_kernel<<<1, 64, 0, stream>>>(sigma, nacc);
  }

  final_scale_kernel<<<(BB * TPR_PER_B) / (256 * 4), 256, 0, stream>>>(T, sigma);
}
